// MMDLoss_26096221290993
// MI455X (gfx1250) — compile-verified, matrix operand order checked by reading, not run on hardware
//
#include <hip/hip_runtime.h>

// ---------------------------------------------------------------------------
// MMD loss on gfx1250 (MI455X): split-bf16 WMMA Gram + fused distance/exp,
// with async global->LDS staging (GLOBAL_LOAD_ASYNC_TO_LDS_B128 / ASYNCcnt).
// ---------------------------------------------------------------------------

typedef __bf16 v8bf  __attribute__((ext_vector_type(8)));
typedef __bf16 v16bf __attribute__((ext_vector_type(16)));
typedef float  v8f   __attribute__((ext_vector_type(8)));

#define N_PTS 8192
#define DIMS  128
#define LDSK  136          // padded LDS row stride (bf16 elems): breaks bank conflicts, keeps 16B align
#define BM    128          // block tile rows (A rows)
#define BN    64           // block tile cols (B rows)

static constexpr size_t SMEM_BYTES =
    (size_t)(2 * BM * LDSK + 2 * BN * LDSK) * 2 /*bf16*/ + 8 * sizeof(float);

union Frag { v16bf v; v8bf h[2]; };

// Async 16B/lane copy: global (SGPR base + 32-bit VGPR offset, GVS mode)
// directly into LDS at a per-lane LDS byte address. Tracked by ASYNCcnt.
static __device__ __forceinline__ void async_ld_b128(void* lds_dst, unsigned long long gbase,
                                                     unsigned int goff_bytes) {
  unsigned int lds_off = (unsigned int)(unsigned long long)(uintptr_t)lds_dst;
  asm volatile("global_load_async_to_lds_b128 %0, %1, %2"
               :: "v"(lds_off), "v"(goff_bytes), "s"(gbase)
               : "memory");
}

static __device__ __forceinline__ void wait_async_zero() {
  asm volatile("s_wait_asynccnt 0" ::: "memory");
}

// Load one 16x32 bf16 WMMA operand fragment from LDS, following the CDNA5
// 16-bit A/B striping: lanes 0-15 hold row (rowBase+lane), K = {kc..kc+7, kc+16..kc+23};
// lanes 16-31 hold row (rowBase+lane-16), K = {kc+8..kc+15, kc+24..kc+31}.
static __device__ __forceinline__ v16bf load_frag_lds(const __bf16* base, int rowBase,
                                                      int lane, int kc) {
  const __bf16* p = base + (rowBase + (lane & 15)) * LDSK + kc + ((lane >> 4) << 3);
  Frag f;
  f.h[0] = *(const v8bf*)(p);
  f.h[1] = *(const v8bf*)(p + 16);
  return f.v;
}

// ---------------------------------------------------------------------------
// Kernel 1: build bf16 hi/lo split of x and y + exact fp32 squared row norms.
// One 128-thread block per row; blocks [0,8192) -> x, [8192,16384) -> y.
// ---------------------------------------------------------------------------
__global__ __launch_bounds__(128) void mmd_prep_kernel(
    const float* __restrict__ x, const float* __restrict__ y,
    __bf16* __restrict__ xhi, __bf16* __restrict__ xlo,
    __bf16* __restrict__ yhi, __bf16* __restrict__ ylo,
    float* __restrict__ xn, float* __restrict__ yn,
    double* __restrict__ sums)
{
  const int b = blockIdx.x;
  const int t = threadIdx.x;
  if (b == 0 && t < 3) sums[t] = 0.0;   // re-zeroed every call: graph-replay safe

  const float* src; __bf16* hi; __bf16* lo; float* nrm; int row;
  if (b < N_PTS) { src = x; hi = xhi; lo = xlo; nrm = xn; row = b; }
  else           { src = y; hi = yhi; lo = ylo; nrm = yn; row = b - N_PTS; }

  const float v  = src[(size_t)row * DIMS + t];
  const __bf16 h = (__bf16)v;
  const __bf16 l = (__bf16)(v - (float)h);
  hi[(size_t)row * DIMS + t] = h;
  lo[(size_t)row * DIMS + t] = l;

  // exact fp32 squared-norm reduction across the row (4 waves of 32)
  float s = v * v;
  #pragma unroll
  for (int off = 16; off > 0; off >>= 1) s += __shfl_down(s, off, 32);
  __shared__ float ws4[4];
  if ((t & 31) == 0) ws4[t >> 5] = s;
  __syncthreads();
  if (t == 0) nrm[row] = ws4[0] + ws4[1] + ws4[2] + ws4[3];
}

// ---------------------------------------------------------------------------
// Kernel 2: one 128x64 tile of kernel matrix z (0: xx, 1: yy, 2: xy).
// 8 waves; each wave owns a 32x32 region = 2x2 WMMA accumulators.
// Gram via split bf16: g = Ahi*Bhi + Ahi*Blo + Alo*Bhi (fp32-class accuracy).
// ---------------------------------------------------------------------------
__global__ __launch_bounds__(256) void mmd_pair_kernel(
    const __bf16* __restrict__ xhi, const __bf16* __restrict__ xlo,
    const __bf16* __restrict__ yhi, const __bf16* __restrict__ ylo,
    const float* __restrict__ xn, const float* __restrict__ yn,
    double* __restrict__ sums)
{
  extern __shared__ char smem[];
  __bf16* AsHi = (__bf16*)smem;
  __bf16* AsLo = AsHi + BM * LDSK;
  __bf16* BsHi = AsLo + BM * LDSK;
  __bf16* BsLo = BsHi + BN * LDSK;
  float*  wsum = (float*)(BsLo + BN * LDSK);

  const int z = blockIdx.z;
  const __bf16* Ahi = (z == 1) ? yhi : xhi;
  const __bf16* Alo = (z == 1) ? ylo : xlo;
  const __bf16* Bhi = (z == 0) ? xhi : yhi;
  const __bf16* Blo = (z == 0) ? xlo : ylo;
  const float*  an  = (z == 1) ? yn : xn;
  const float*  bn  = (z == 0) ? xn : yn;

  const unsigned long long aHiB = (unsigned long long)(uintptr_t)Ahi;
  const unsigned long long aLoB = (unsigned long long)(uintptr_t)Alo;
  const unsigned long long bHiB = (unsigned long long)(uintptr_t)Bhi;
  const unsigned long long bLoB = (unsigned long long)(uintptr_t)Blo;

  const int mBase = blockIdx.y * BM;
  const int nBase = blockIdx.x * BN;
  const int tid   = threadIdx.x;

  // ---- async global -> LDS staging: 16B per lane, no VGPR round-trip ----
  for (int c = tid; c < (BM * DIMS) / 8; c += 256) {
    const int row = c >> 4;            // 16 chunks of 8 bf16 per 128-elem row
    const int col = (c & 15) << 3;
    const unsigned int g = ((unsigned int)(mBase + row) * DIMS + col) * 2u; // bytes
    async_ld_b128(&AsHi[row * LDSK + col], aHiB, g);
    async_ld_b128(&AsLo[row * LDSK + col], aLoB, g);
  }
  for (int c = tid; c < (BN * DIMS) / 8; c += 256) {
    const int row = c >> 4;
    const int col = (c & 15) << 3;
    const unsigned int g = ((unsigned int)(nBase + row) * DIMS + col) * 2u; // bytes
    async_ld_b128(&BsHi[row * LDSK + col], bHiB, g);
    async_ld_b128(&BsLo[row * LDSK + col], bLoB, g);
  }
  wait_async_zero();
  __syncthreads();

  const int wave = tid >> 5;
  const int lane = tid & 31;
  const int mOff = (wave >> 1) * 32;   // 0,32,64,96
  const int nOff = (wave & 1) * 32;    // 0,32

  v8f acc[2][2] = {};

  #pragma unroll
  for (int kc = 0; kc < DIMS; kc += 32) {
    v16bf aHi[2], aLo[2], bHiF[2], bLoF[2];
    #pragma unroll
    for (int mt = 0; mt < 2; ++mt) {
      aHi[mt] = load_frag_lds(AsHi, mOff + mt * 16, lane, kc);
      aLo[mt] = load_frag_lds(AsLo, mOff + mt * 16, lane, kc);
    }
    #pragma unroll
    for (int nt = 0; nt < 2; ++nt) {
      bHiF[nt] = load_frag_lds(BsHi, nOff + nt * 16, lane, kc);
      bLoF[nt] = load_frag_lds(BsLo, nOff + nt * 16, lane, kc);
    }
    #pragma unroll
    for (int mt = 0; mt < 2; ++mt)
      #pragma unroll
      for (int nt = 0; nt < 2; ++nt) {
        acc[mt][nt] = __builtin_amdgcn_wmma_f32_16x16x32_bf16(
            false, aHi[mt], false, bHiF[nt], (short)0, acc[mt][nt], false, false);
        acc[mt][nt] = __builtin_amdgcn_wmma_f32_16x16x32_bf16(
            false, aHi[mt], false, bLoF[nt], (short)0, acc[mt][nt], false, false);
        acc[mt][nt] = __builtin_amdgcn_wmma_f32_16x16x32_bf16(
            false, aLo[mt], false, bHiF[nt], (short)0, acc[mt][nt], false, false);
      }
  }

  // ---- fused epilogue: d^2 = |a|^2 + |b|^2 - 2g ; exp(-0.5*sqrt(d^2)) ----
  // C/D layout: VGPR v -> M = v + 8*(lane>=16); N = lane & 15.
  float lsum = 0.0f;
  #pragma unroll
  for (int mt = 0; mt < 2; ++mt)
    #pragma unroll
    for (int nt = 0; nt < 2; ++nt) {
      const int n  = nBase + nOff + nt * 16 + (lane & 15);
      const float nb = bn[n];
      const int m0 = mBase + mOff + mt * 16 + ((lane >> 4) << 3);
      #pragma unroll
      for (int v = 0; v < 8; ++v) {
        float d2 = an[m0 + v] + nb - 2.0f * acc[mt][nt][v];
        d2 = fmaxf(d2, 0.0f);                 // safe sqrt; diagonal -> exp(0)=1
        lsum += __expf(-0.5f * sqrtf(d2));
      }
    }

  // wave32 reduce, then block reduce, one f64 atomic per block
  #pragma unroll
  for (int off = 16; off > 0; off >>= 1) lsum += __shfl_down(lsum, off, 32);
  if (lane == 0) wsum[wave] = lsum;
  __syncthreads();
  if (tid == 0) {
    float bsum = 0.0f;
    #pragma unroll
    for (int i = 0; i < 8; ++i) bsum += wsum[i];
    atomicAdd(&sums[z], (double)bsum);
  }
}

// ---------------------------------------------------------------------------
// Kernel 3: combine the three sums into the scalar loss.
// ---------------------------------------------------------------------------
__global__ void mmd_finalize_kernel(const double* __restrict__ sums, float* __restrict__ out) {
  if (threadIdx.x == 0 && blockIdx.x == 0) {
    const double inv = 1.0 / ((double)N_PTS * (double)N_PTS);
    out[0] = (float)((sums[0] + sums[1] - 2.0 * sums[2]) * inv);
  }
}

// ---------------------------------------------------------------------------
extern "C" void kernel_launch(void* const* d_in, const int* in_sizes, int n_in,
                              void* d_out, int out_size, void* d_ws, size_t ws_size,
                              hipStream_t stream) {
  const float* x = (const float*)d_in[0];
  const float* y = (const float*)d_in[1];

  char* ws = (char*)d_ws;
  const size_t MAT = (size_t)N_PTS * DIMS;          // 1,048,576 elems
  __bf16* xhi = (__bf16*)(ws);
  __bf16* xlo = (__bf16*)(ws + 2 * MAT);
  __bf16* yhi = (__bf16*)(ws + 4 * MAT);
  __bf16* ylo = (__bf16*)(ws + 6 * MAT);
  float*  xn  = (float*)(ws + 8 * MAT);
  float*  yn  = (float*)(ws + 8 * MAT + (size_t)N_PTS * 4);
  double* sums = (double*)(ws + 8 * MAT + (size_t)2 * N_PTS * 4);

  mmd_prep_kernel<<<2 * N_PTS, 128, 0, stream>>>(x, y, xhi, xlo, yhi, ylo, xn, yn, sums);

  dim3 grid(N_PTS / BN, N_PTS / BM, 3);             // (128, 64, 3)
  mmd_pair_kernel<<<grid, 256, SMEM_BYTES, stream>>>(xhi, xlo, yhi, ylo, xn, yn, sums);

  mmd_finalize_kernel<<<1, 32, 0, stream>>>(sums, (float*)d_out);
}
